// WaveNeuronLayer_6330781794736
// MI455X (gfx1250) — matrix operand order repaired and checked by reading, not compiled, not run on hardware
//
#include <hip/hip_runtime.h>
#include <math.h>

// ---------------------------------------------------------------------------
// WaveNeuronLayer for MI455X (gfx1250, wave32, WMMA bf16 16x16x32)
//
// Pipeline (all on `stream`):
//   k_pack_w1 / k_pack_w4 / k_setup : pre-swizzle weights into CDNA5 WMMA
//       fragment layouts (bf16), compute coupled[16], zero global sums.
//   k_wave_in  : x@w1 (bf16 WMMA, fp32 acc) -> LN -> SiLU -> @w2 ->
//                modulated (ws), partial sums for mean_waves (atomics).
//   k_factor   : resonance_factor scalar.
//   k_wave_out : modulated@w3 (WMMA, K zero-padded) -> LN -> SiLU ->
//                @w4 (WMMA) -> out = x + factor*(...)   [fused residual]
//
// Workspace layout (bytes), total ~1.50 MiB:
//   [0,64)        sums[16]      f32
//   [64,128)      coupled[16]   f32
//   [128,132)     factor        f32
//   [256, +256KB) w1 fragments  bf16 (128 kblocks x 2 ntiles x 512)
//   then 2KB      w3 fragments  bf16 (2 ntiles x 512, K padded 16->32)
//   then 256KB    w4 fragments  bf16 (256 ntiles x 512)
//   then 1MB      modulated     f32  (16384 rows x 16)
// ---------------------------------------------------------------------------

#define HID 4096
#define WW  16

typedef __attribute__((ext_vector_type(16))) __bf16 bf16x16;
typedef __attribute__((ext_vector_type(8)))  __bf16 bf16x8;
typedef __attribute__((ext_vector_type(8)))  float  f32x8;

static __device__ __forceinline__ float sigmoidf_(float v) {
    return 1.0f / (1.0f + __expf(-v));
}

// ------------------------- weight pre-swizzle kernels ----------------------
// A-fragment (16-bit, 16x32 MxK), lane L holds row M=L%16:
//   koff=(L>=16)*8 ; slots 0..7 -> K=koff+s ; slots 8..15 -> K=16+koff+(s-8)
// B-fragment (16-bit, 32x16 KxN), lane L holds col N=L%16:
//   koff=(L>=16)*8 ; slots 0..7 -> K=koff+s ; slots 8..15 -> K=16+koff+(s-8)
// Fragments stored as 512 contiguous bf16: frag*512 + lane*16 + slot.

__global__ __launch_bounds__(256) void k_pack_w1(const float* __restrict__ w1,
                                                 __bf16* __restrict__ w1f) {
    int e = blockIdx.x * 256 + threadIdx.x;          // 131072 elements
    int frag = e >> 9, within = e & 511;
    int lane = within >> 4, slot = within & 15;
    int kb = frag >> 1, nt = frag & 1;
    int kbase = (lane >> 4) * 8 + (slot >> 3) * 8;   // B layout
    int k = kb * 32 + kbase + slot;
    int n = nt * 16 + (lane & 15);
    w1f[e] = (__bf16)w1[k * 32 + n];
}

__global__ __launch_bounds__(256) void k_pack_w4(const float* __restrict__ w4,
                                                 __bf16* __restrict__ w4f) {
    int e = blockIdx.x * 256 + threadIdx.x;          // 131072 elements
    int nt = e >> 9, within = e & 511;
    int lane = within >> 4, slot = within & 15;
    int k = (lane >> 4) * 8 + (slot >> 3) * 8 + slot; // B layout, K=32
    int n = nt * 16 + (lane & 15);
    w4f[e] = (__bf16)w4[(size_t)k * HID + n];
}

__global__ __launch_bounds__(256) void k_setup(const float* __restrict__ freq,
                                               const float* __restrict__ ph,
                                               const float* __restrict__ cpl,
                                               const float* __restrict__ damp,
                                               const float* __restrict__ w3,
                                               const float* __restrict__ tstep,
                                               float* __restrict__ sums,
                                               float* __restrict__ coupled,
                                               __bf16* __restrict__ w3f) {
    int tid = threadIdx.x;
    if (tid < WW) {
        sums[tid] = 0.0f;
        float t = tstep[0] + 0.1f;
        float acc = 0.0f;
        for (int i = 0; i < WW; ++i) {
            float bw = __sinf(6.283185307179586f * freq[i] * t + ph[i]) *
                       __expf(-damp[i] * t);
            acc += bw * cpl[i * WW + tid];
        }
        coupled[tid] = tanhf(acc);
    }
    // w3 fragments: 2 ntiles x 512, K padded 16 -> 32 with zeros
    for (int e = tid; e < 1024; e += 256) {
        int frag = e >> 9, within = e & 511;
        int lane = within >> 4, slot = within & 15;
        int k = (lane >> 4) * 8 + (slot >> 3) * 8 + slot;
        int n = frag * 16 + (lane & 15);
        w3f[e] = (k < WW) ? (__bf16)w3[k * 32 + n] : (__bf16)0.0f;
    }
}

// ------------------------------ phase 1 ------------------------------------
__global__ __launch_bounds__(256) void k_wave_in(
    const float* __restrict__ x, const float* __restrict__ b1,
    const float* __restrict__ g1, const float* __restrict__ be1,
    const float* __restrict__ w2, const float* __restrict__ b2,
    const float* __restrict__ coupled, const __bf16* __restrict__ w1f,
    float* __restrict__ modw, float* __restrict__ gsums) {
    __shared__ float stage[8][512];                  // 16 rows x 32 per wave
    __shared__ float acc[WW];
    __shared__ float s_w2[512];
    __shared__ float s_g1[32], s_be1[32], s_b2[WW], s_cpl[WW];

    int tid = threadIdx.x;
    int wave = tid >> 5, lane = tid & 31;
    int n = lane & 15, hi = lane >> 4;

    s_w2[tid] = w2[tid];
    s_w2[tid + 256] = w2[tid + 256];
    if (tid < 32) { s_g1[tid] = g1[tid]; s_be1[tid] = be1[tid]; }
    if (tid < WW) { s_b2[tid] = b2[tid]; s_cpl[tid] = coupled[tid]; acc[tid] = 0.0f; }

    int tile = blockIdx.x * 8 + wave;
    int row0 = tile * 16;
    const float* xrow = x + (size_t)(row0 + n) * HID + hi * 8;   // koff fold

    f32x8 c0, c1;
#pragma unroll
    for (int r = 0; r < 8; ++r) { c0[r] = 0.0f; c1[r] = 0.0f; }

    const bf16x16* bp = (const bf16x16*)w1f;   // 1 fragment = 32 bf16x16 units
    for (int kb = 0; kb < 128; ++kb) {
        const float4* pa0 = (const float4*)(xrow + kb * 32);
        const float4* pa1 = (const float4*)(xrow + kb * 32 + 16);
        float4 a0 = pa0[0], a1 = pa0[1];       // K = kb*32 + koff + 0..7
        float4 a2 = pa1[0], a3 = pa1[1];       // K = kb*32 + 16 + koff + 0..7
        bf16x16 A;
        A[0]  = (__bf16)a0.x; A[1]  = (__bf16)a0.y; A[2]  = (__bf16)a0.z; A[3]  = (__bf16)a0.w;
        A[4]  = (__bf16)a1.x; A[5]  = (__bf16)a1.y; A[6]  = (__bf16)a1.z; A[7]  = (__bf16)a1.w;
        A[8]  = (__bf16)a2.x; A[9]  = (__bf16)a2.y; A[10] = (__bf16)a2.z; A[11] = (__bf16)a2.w;
        A[12] = (__bf16)a3.x; A[13] = (__bf16)a3.y; A[14] = (__bf16)a3.z; A[15] = (__bf16)a3.w;
        bf16x16 B0 = bp[(kb * 2 + 0) * 32 + lane];
        bf16x16 B1 = bp[(kb * 2 + 1) * 32 + lane];
        c0 = __builtin_amdgcn_wmma_f32_16x16x32_bf16(false, A, false, B0,
                                                     (short)0, c0, false, false);
        c1 = __builtin_amdgcn_wmma_f32_16x16x32_bf16(false, A, false, B1,
                                                     (short)0, c1, false, false);
    }

    float b1lo = b1[n], b1hi = b1[16 + n];
#pragma unroll
    for (int r = 0; r < 8; ++r) {
        stage[wave][(r + 8 * hi) * 32 + n]      = c0[r] + b1lo;
        stage[wave][(r + 8 * hi) * 32 + 16 + n] = c1[r] + b1hi;
    }
    __syncthreads();

    if (lane < 16) {
        int row = row0 + lane;
        const float* h = &stage[wave][lane * 32];
        float mu = 0.0f;
#pragma unroll
        for (int d = 0; d < 32; ++d) mu += h[d];
        mu *= (1.0f / 32.0f);
        float var = 0.0f;
#pragma unroll
        for (int d = 0; d < 32; ++d) { float df = h[d] - mu; var += df * df; }
        var *= (1.0f / 32.0f);
        float inv = rsqrtf(var + 1e-5f);
        float s[32];
#pragma unroll
        for (int d = 0; d < 32; ++d) {
            float v = (h[d] - mu) * inv * s_g1[d] + s_be1[d];
            s[d] = v * sigmoidf_(v);                 // SiLU
        }
        for (int w = 0; w < WW; ++w) {
            float wi = s_b2[w];
#pragma unroll
            for (int d = 0; d < 32; ++d) wi += s[d] * s_w2[d * WW + w];
            float mv = s_cpl[w] * (1.0f + wi);       // modulated
            modw[(size_t)row * WW + w] = mv;
            atomicAdd(&acc[w], mv);
        }
    }
    __syncthreads();
    if (tid < WW) atomicAdd(&gsums[tid], acc[tid]);
}

// ------------------------------ phase 2 ------------------------------------
__global__ void k_factor(const float* __restrict__ sums,
                         float* __restrict__ factor, float count) {
    if (threadIdx.x == 0 && blockIdx.x == 0) {
        float m = 0.0f;
        for (int w = 0; w < WW; ++w) m += fabsf(sums[w]);
        m = 0.1f * m / (count * (float)WW);          // mean(0.1*|mean_waves|)
        factor[0] = sigmoidf_(m);
    }
}

// ------------------------------ phase 3 ------------------------------------
__global__ __launch_bounds__(256) void k_wave_out(
    const float* __restrict__ x, const float* __restrict__ b3,
    const float* __restrict__ g2, const float* __restrict__ be2,
    const float* __restrict__ b4, const float* __restrict__ modw,
    const float* __restrict__ factorp, const __bf16* __restrict__ w3f,
    const __bf16* __restrict__ w4f, float* __restrict__ out) {
    __shared__ float  stage[8][512];
    __shared__ __bf16 sa[8][512];
    __shared__ float  s_g2[32], s_be2[32];

    int tid = threadIdx.x;
    int wave = tid >> 5, lane = tid & 31;
    int n = lane & 15, hi = lane >> 4;
    if (tid < 32) { s_g2[tid] = g2[tid]; s_be2[tid] = be2[tid]; }

    int tile = blockIdx.x * 8 + wave;
    int row0 = tile * 16;
    int koff = hi * 8;

    // A3 fragment from modulated (K=16 real, K=16..31 zero-padded)
    const float4* mp = (const float4*)(modw + (size_t)(row0 + n) * WW + koff);
    float4 m0 = mp[0], m1 = mp[1];
    bf16x16 A3;
    A3[0] = (__bf16)m0.x; A3[1] = (__bf16)m0.y; A3[2] = (__bf16)m0.z; A3[3] = (__bf16)m0.w;
    A3[4] = (__bf16)m1.x; A3[5] = (__bf16)m1.y; A3[6] = (__bf16)m1.z; A3[7] = (__bf16)m1.w;
#pragma unroll
    for (int i = 8; i < 16; ++i) A3[i] = (__bf16)0.0f;

    const bf16x16* b3p = (const bf16x16*)w3f;
    bf16x16 B30 = b3p[lane], B31 = b3p[32 + lane];
    float bb0 = b3[n], bb1 = b3[16 + n];
    f32x8 c0, c1;
#pragma unroll
    for (int r = 0; r < 8; ++r) { c0[r] = bb0; c1[r] = bb1; }
    c0 = __builtin_amdgcn_wmma_f32_16x16x32_bf16(false, A3, false, B30,
                                                 (short)0, c0, false, false);
    c1 = __builtin_amdgcn_wmma_f32_16x16x32_bf16(false, A3, false, B31,
                                                 (short)0, c1, false, false);

#pragma unroll
    for (int r = 0; r < 8; ++r) {
        stage[wave][(r + 8 * hi) * 32 + n]      = c0[r];
        stage[wave][(r + 8 * hi) * 32 + 16 + n] = c1[r];
    }
    __syncthreads();

    if (lane < 16) {
        const float* o = &stage[wave][lane * 32];
        float mu = 0.0f;
#pragma unroll
        for (int d = 0; d < 32; ++d) mu += o[d];
        mu *= (1.0f / 32.0f);
        float var = 0.0f;
#pragma unroll
        for (int d = 0; d < 32; ++d) { float df = o[d] - mu; var += df * df; }
        var *= (1.0f / 32.0f);
        float inv = rsqrtf(var + 1e-5f);
#pragma unroll
        for (int d = 0; d < 32; ++d) {
            float v = (o[d] - mu) * inv * s_g2[d] + s_be2[d];
            v = v * sigmoidf_(v);                    // SiLU
            sa[wave][lane * 32 + d] = (__bf16)v;
        }
    }
    __syncthreads();

    // A4 fragment (K=32) from LDS bf16 staging, two ds b128 reads per lane
    const bf16x8* sap = (const bf16x8*)&sa[wave][n * 32];
    bf16x8 alo = sap[koff >> 3];                     // K = koff + 0..7
    bf16x8 ahi = sap[(16 + koff) >> 3];              // K = 16 + koff + 0..7
    bf16x16 A4;
#pragma unroll
    for (int i = 0; i < 8; ++i) { A4[i] = alo[i]; A4[8 + i] = ahi[i]; }

    float factor = factorp[0];
    const bf16x16* w4p = (const bf16x16*)w4f;

    for (int nt = 0; nt < 256; ++nt) {
        bf16x16 B4 = w4p[nt * 32 + lane];
        float cb = b4[nt * 16 + n];
        f32x8 c;
#pragma unroll
        for (int r = 0; r < 8; ++r) c[r] = cb;
        c = __builtin_amdgcn_wmma_f32_16x16x32_bf16(false, A4, false, B4,
                                                    (short)0, c, false, false);
#pragma unroll
        for (int r = 0; r < 8; ++r) {
            size_t idx = (size_t)(row0 + r + 8 * hi) * HID + nt * 16 + n;
            out[idx] = x[idx] + factor * c[r];       // fused residual
        }
    }
}

// ------------------------------- launcher ----------------------------------
extern "C" void kernel_launch(void* const* d_in, const int* in_sizes, int n_in,
                              void* d_out, int out_size, void* d_ws, size_t ws_size,
                              hipStream_t stream) {
    const float* x     = (const float*)d_in[0];
    const float* freq  = (const float*)d_in[1];
    const float* ph    = (const float*)d_in[2];
    const float* cpl   = (const float*)d_in[3];
    const float* damp  = (const float*)d_in[4];
    const float* w1    = (const float*)d_in[5];
    const float* b1    = (const float*)d_in[6];
    const float* g1    = (const float*)d_in[7];
    const float* be1   = (const float*)d_in[8];
    const float* w2    = (const float*)d_in[9];
    const float* b2    = (const float*)d_in[10];
    const float* w3    = (const float*)d_in[11];
    const float* b3    = (const float*)d_in[12];
    const float* g2    = (const float*)d_in[13];
    const float* be2   = (const float*)d_in[14];
    const float* w4    = (const float*)d_in[15];
    const float* b4    = (const float*)d_in[16];
    const float* tstep = (const float*)d_in[17];
    float* out = (float*)d_out;

    char* ws = (char*)d_ws;
    float*  sums    = (float*)(ws + 0);
    float*  coupled = (float*)(ws + 64);
    float*  factor  = (float*)(ws + 128);
    __bf16* w1f     = (__bf16*)(ws + 256);
    __bf16* w3f     = (__bf16*)(ws + 256 + 262144);
    __bf16* w4f     = (__bf16*)(ws + 256 + 262144 + 2048);
    float*  modw    = (float*)(ws + 256 + 262144 + 2048 + 262144);

    int rows = in_sizes[0] / HID;                    // 16384 = B*S

    k_pack_w1<<<512, 256, 0, stream>>>(w1, w1f);
    k_pack_w4<<<512, 256, 0, stream>>>(w4, w4f);
    k_setup<<<1, 256, 0, stream>>>(freq, ph, cpl, damp, w3, tstep,
                                   sums, coupled, w3f);
    k_wave_in<<<rows / 128, 256, 0, stream>>>(x, b1, g1, be1, w2, b2,
                                              coupled, w1f, modw, sums);
    k_factor<<<1, 32, 0, stream>>>(sums, factor, (float)rows);
    k_wave_out<<<rows / 128, 256, 0, stream>>>(x, b3, g2, be2, b4, modw,
                                               factor, w3f, w4f, out);
}
